// CrossProduct_18056042513139
// MI455X (gfx1250) — matrix operand order repaired and checked by reading, not compiled, not run on hardware
//
#include <hip/hip_runtime.h>

typedef __attribute__((ext_vector_type(16))) _Float16 v16h;
typedef __attribute__((ext_vector_type(8)))  float    v8f;
typedef __attribute__((ext_vector_type(4)))  float    v4f;

#define NB 16384
#define XD 1024
#define KO 64

// Workspace layout:
//   [0, 131072)       : V packed to fp16 in WMMA B-operand lane layout
//                       flat = ((kb*4 + t)*32 + lane)*16 + e
//                       K = kb*32 + 8*(lane>=16) + (e&7) + 16*(e>=8) ; N = t*16 + (lane&15)
//   [131072, 135168)  : s[1024], s_i = sum_k V[i,k]^2  (fp32)

__global__ __launch_bounds__(256) void fm_prep(const float* __restrict__ v,
                                               _Float16* __restrict__ bp,
                                               float* __restrict__ s) {
  int tid  = blockIdx.x * 256 + threadIdx.x;     // 0..65535
  int e    = tid & 15;
  int lane = (tid >> 4) & 31;
  int t    = (tid >> 9) & 3;
  int kb   = tid >> 11;                          // 0..31
  int kg   = kb * 32 + ((lane >> 4) << 3) + (e & 7) + ((e >> 3) << 4);   // GEMM K index (x-dim)
  int n    = t * 16 + (lane & 15);               // output column
  bp[tid] = (_Float16)v[kg * KO + n];
  if (tid < XD) {
    float acc = 0.f;
    const float* vr = v + tid * KO;
#pragma unroll 8
    for (int k = 0; k < KO; ++k) acc += vr[k] * vr[k];
    s[tid] = acc;
  }
}

__global__ __launch_bounds__(128) void fm_main(const float* __restrict__ x,
                                               const _Float16* __restrict__ bp,
                                               const float* __restrict__ sg,
                                               float* __restrict__ out) {
  __shared__ float s_lds[XD];
  __shared__ float red_sq[4][16];
  __shared__ float red_s2[4][16];

  const int tid = threadIdx.x;
  for (int i = tid; i < XD; i += 128) s_lds[i] = sg[i];
  __syncthreads();

  const int wave    = tid >> 5;
  const int lane    = tid & 31;
  const int rowBase = blockIdx.x * 64 + wave * 16;
  const int row     = rowBase + (lane & 15);
  const int off     = (lane >> 4) << 3;          // 0 or 8 (A-layout K offset)

  const float* xrow = x + (size_t)row * XD;
  const v16h*  bpv  = (const v16h*)bp + (size_t)lane;

  v8f c0 = {}, c1 = {}, c2 = {}, c3 = {};
  v4f s2a = {}, s2b = {};                        // vector fp32 accumulators for sum(x^2 * s)

#pragma unroll 2
  for (int kb = 0; kb < 32; ++kb) {
    const int k0 = kb * 32 + off;
    // A tile: row `row`, K = k0+0..7 and k0+16..23 (per-lane ISA layout)
    v4f xa0 = __builtin_nontemporal_load((const v4f*)(xrow + k0));
    v4f xa1 = __builtin_nontemporal_load((const v4f*)(xrow + k0 + 4));
    v4f xb0 = __builtin_nontemporal_load((const v4f*)(xrow + k0 + 16));
    v4f xb1 = __builtin_nontemporal_load((const v4f*)(xrow + k0 + 20));

    v4f sa0 = *(const v4f*)(s_lds + k0);
    v4f sa1 = *(const v4f*)(s_lds + k0 + 4);
    v4f sb0 = *(const v4f*)(s_lds + k0 + 16);
    v4f sb1 = *(const v4f*)(s_lds + k0 + 20);

    // sum(x^2 * s): aligned quads -> v_pk_mul_f32 + v_pk_fma_f32, no reg shuffling
    v4f t0 = xa0 * xa0;
    v4f t1 = xa1 * xa1;
    v4f t2 = xb0 * xb0;
    v4f t3 = xb1 * xb1;
    s2a = t0 * sa0 + s2a;
    s2b = t1 * sa1 + s2b;
    s2a = t2 * sb0 + s2a;
    s2b = t3 * sb1 + s2b;

    v16h a;
#pragma unroll
    for (int j = 0; j < 4; ++j) {
      a[j]      = (_Float16)xa0[j];
      a[4 + j]  = (_Float16)xa1[j];
      a[8 + j]  = (_Float16)xb0[j];
      a[12 + j] = (_Float16)xb1[j];
    }

    const v16h* bk = bpv + (size_t)kb * 128;     // (kb*4+t)*32 + lane
    v16h b0 = bk[0];
    v16h b1 = bk[32];
    v16h b2 = bk[64];
    v16h b3 = bk[96];

    c0 = __builtin_amdgcn_wmma_f32_16x16x32_f16(false, a, false, b0, (short)0, c0, false, false);
    c1 = __builtin_amdgcn_wmma_f32_16x16x32_f16(false, a, false, b1, (short)0, c1, false, false);
    c2 = __builtin_amdgcn_wmma_f32_16x16x32_f16(false, a, false, b2, (short)0, c2, false, false);
    c3 = __builtin_amdgcn_wmma_f32_16x16x32_f16(false, a, false, b3, (short)0, c3, false, false);
  }

  // horizontal reduce of the vector sum2 accumulators
  v4f s2v = s2a + s2b;
  float sum2 = (s2v[0] + s2v[1]) + (s2v[2] + s2v[3]);

  // per-lane partial sum over its 4 N-columns, per C/D row r (M = r + 8*(lane>=16))
  float p[8];
#pragma unroll
  for (int r = 0; r < 8; ++r)
    p[r] = c0[r] * c0[r] + c1[r] * c1[r] + c2[r] * c2[r] + c3[r] * c3[r];

  // butterfly-reduce over the 16 lanes of each half (covers all N=0..63)
#pragma unroll
  for (int m = 1; m <= 8; m <<= 1) {
#pragma unroll
    for (int r = 0; r < 8; ++r) p[r] += __shfl_xor(p[r], m, 32);
  }
  // combine the two K-halves of sum(x^2 * s) for this row
  sum2 += __shfl_xor(sum2, 16, 32);

  if ((lane & 15) == 0) {
    const int mb = (lane >> 4) * 8;              // rows 0..7 (half0) / 8..15 (half1)
#pragma unroll
    for (int r = 0; r < 8; ++r) red_sq[wave][mb + r] = p[r];
  }
  if (lane < 16) red_s2[wave][lane] = sum2;
  __syncthreads();

  if (lane < 16)
    out[rowBase + lane] = 0.5f * (red_sq[wave][lane] - red_s2[wave][lane]);
}

extern "C" void kernel_launch(void* const* d_in, const int* in_sizes, int n_in,
                              void* d_out, int out_size, void* d_ws, size_t ws_size,
                              hipStream_t stream) {
  const float* x = (const float*)d_in[0];   // (16384, 1024) fp32
  const float* v = (const float*)d_in[1];   // (1024, 64)    fp32
  float* out = (float*)d_out;               // (16384, 1)    fp32
  _Float16* bp = (_Float16*)d_ws;
  float* s = (float*)((char*)d_ws + 131072);

  fm_prep<<<256, 256, 0, stream>>>(v, bp, s);
  fm_main<<<256, 128, 0, stream>>>(x, bp, s, out);
}
